// DGCNN_geo_3513283248947
// MI455X (gfx1250) — matrix operand order repaired and checked by reading, not compiled, not run on hardware
//
#include <hip/hip_runtime.h>
#include <hip/hip_bf16.h>
#include <cstdint>
#include <cstddef>

// ---------------------------------------------------------------------------
// Types for CDNA5 WMMA (wave32) + TDM descriptor groups
// ---------------------------------------------------------------------------
typedef __attribute__((ext_vector_type(16))) __bf16 bf16x16;
typedef __attribute__((ext_vector_type(8)))  float  floatx8;
typedef __attribute__((ext_vector_type(8)))  unsigned uintx8;
typedef __attribute__((ext_vector_type(4)))  unsigned uint32x4;
typedef __attribute__((ext_vector_type(8)))  int      int32x8;
typedef __attribute__((ext_vector_type(4)))  int      int32x4;

#define NPTS   2048
#define KNBR   20
#define BATCH  8
#define NEG_SLOPE 0.2f
#define BN_EPS 1e-5f

#if defined(__HIP_DEVICE_COMPILE__) && defined(__gfx1250__) && \
    __has_builtin(__builtin_amdgcn_tensor_load_to_lds)
#define HAVE_TDM 1
#else
#define HAVE_TDM 0
#endif

// ---------------------------------------------------------------------------
// TDM: 2-D tile load Global -> LDS (D# per CDNA5 ISA).
// data_size_code: 0=1B 1=2B 2=4B 3=8B. Strides/dims in data_size units.
// Issue from ONE wave only (EXEC ignored; one issue per executing wave);
// that wave must s_wait_tensorcnt(0) before the workgroup barrier.
// This toolchain's builtin takes 6 args (g0, g1, g2, g3, g4-spare, cpol).
// ---------------------------------------------------------------------------
#if HAVE_TDM
__device__ __forceinline__ void tdm_load_2d(const void* gaddr, unsigned lds_off,
                                            unsigned tensor_d0, unsigned tensor_d1,
                                            unsigned tile_d0, unsigned tile_d1,
                                            unsigned long long stride0,
                                            unsigned data_size_code,
                                            unsigned pad_enable,
                                            unsigned pad_interval,
                                            unsigned pad_amount) {
  unsigned long long ga = (unsigned long long)(uintptr_t)gaddr;
  uint32x4 g0;
  g0[0] = 1u;                                        // count=1, user mode
  g0[1] = lds_off;                                   // LDS byte address
  g0[2] = (unsigned)(ga & 0xFFFFFFFFu);              // global_addr[31:0]
  g0[3] = (unsigned)((ga >> 32) & 0x01FFFFFFu) | (2u << 30);  // addr[56:32]|type=2
  unsigned w0 = (data_size_code << 16) | (pad_enable << 20) |
                (pad_interval << 22) | (pad_amount << 25);     // wg_mask=0
  unsigned w1 = (tensor_d0 & 0xFFFFu) << 16;                   // tensor_dim0 lo
  unsigned w2 = (tensor_d0 >> 16) | ((tensor_d1 & 0xFFFFu) << 16);
  unsigned w3 = (tensor_d1 >> 16) | (tile_d0 << 16);
  unsigned w4 = tile_d1;                                       // tile_dim2 = 0
  unsigned w5 = (unsigned)(stride0 & 0xFFFFFFFFull);
  unsigned w6 = (unsigned)((stride0 >> 32) & 0xFFFFull);       // dim1_stride = 0
  int32x8 g1 = {(int)w0, (int)w1, (int)w2, (int)w3,
                (int)w4, (int)w5, (int)w6, 0};
  int32x4 z4 = {0, 0, 0, 0};
  int32x8 z8 = {0, 0, 0, 0, 0, 0, 0, 0};
  __builtin_amdgcn_tensor_load_to_lds(g0, g1, z4, z4, z8, 0);
}
__device__ __forceinline__ unsigned lds_offset_of(const void* p) {
  // Generic LDS addresses carry the LDS offset in the low 32 bits.
  return (unsigned)(uintptr_t)p;
}
#endif

// ---------------------------------------------------------------------------
// WMMA fragment loaders (contiguous k order: element s of lane half h holds k = 16h + s).
// ---------------------------------------------------------------------------
__device__ __forceinline__ bf16x16 load_a_frag(const __bf16* base, int stride_elems,
                                               int kk, int lane) {
  const unsigned* p32 =
      reinterpret_cast<const unsigned*>(base + (size_t)(lane & 15) * stride_elems);
  const int g = (lane >> 4) & 1;
  uintx8 t;
#pragma unroll
  for (int v = 0; v < 8; ++v) {
    int kb = kk + (g << 4) + (v << 1);
    t[v] = p32[kb >> 1];
  }
  return __builtin_bit_cast(bf16x16, t);
}

__device__ __forceinline__ bf16x16 load_b_frag(const __bf16* W, int ck,
                                               int o0, int kk, int lane) {
  const int col = lane & 15, g = (lane >> 4) & 1;
  const unsigned* p32 =
      reinterpret_cast<const unsigned*>(W + (size_t)(o0 + col) * ck) +
      ((kk + (g << 4)) >> 1);
  uintx8 t;
#pragma unroll
  for (int j = 0; j < 8; ++j) t[j] = p32[j];
  return __builtin_bit_cast(bf16x16, t);
}

__device__ __forceinline__ unsigned fenc(float x) {
  unsigned u = __float_as_uint(x);
  return (u & 0x80000000u) ? ~u : (u | 0x80000000u);
}
__device__ __forceinline__ float fdec(unsigned e) {
  unsigned u = (e & 0x80000000u) ? (e ^ 0x80000000u) : ~e;
  return __uint_as_float(u);
}

// ---------------------------------------------------------------------------
// Kernel 1: top-k (k=20, descending, tie -> smaller index) per (b,n) row.
// ---------------------------------------------------------------------------
__global__ __launch_bounds__(256)
void topk_kernel(const float* __restrict__ geod, int* __restrict__ idx) {
  __shared__ float vals[NPTS];
  __shared__ float rv[256];
  __shared__ int   ri[256];
  const int row = blockIdx.x;
  const int tid = threadIdx.x;
  const float* src = geod + (size_t)row * NPTS;
  for (int i = tid; i < NPTS; i += 256) vals[i] = src[i];
  __syncthreads();
  for (int kk = 0; kk < KNBR; ++kk) {
    float bv = -3.4e38f; int bi = NPTS;
    for (int i = tid; i < NPTS; i += 256) {
      float v = vals[i];
      if (v > bv || (v == bv && i < bi)) { bv = v; bi = i; }
    }
    rv[tid] = bv; ri[tid] = bi;
    __syncthreads();
    for (int s = 128; s > 0; s >>= 1) {
      if (tid < s) {
        float v2 = rv[tid + s]; int i2 = ri[tid + s];
        if (v2 > rv[tid] || (v2 == rv[tid] && i2 < ri[tid])) { rv[tid] = v2; ri[tid] = i2; }
      }
      __syncthreads();
    }
    if (tid == 0) {
      idx[(size_t)row * KNBR + kk] = ri[0];
      vals[ri[0]] = -3.4e38f;
    }
    __syncthreads();
  }
}

// ---------------------------------------------------------------------------
// Kernel 2: convert f32 weights (O x IC) -> bf16 padded (O x CK); misc zero.
// ---------------------------------------------------------------------------
__global__ void convw_kernel(const float* __restrict__ W, __bf16* __restrict__ Wb,
                             int O, int IC, int CK) {
  int i = blockIdx.x * blockDim.x + threadIdx.x;
  if (i >= O * CK) return;
  int o = i / CK, c = i - o * CK;
  Wb[i] = (c < IC) ? (__bf16)W[(size_t)o * IC + c] : (__bf16)0.0f;
}

__global__ void zero_f32(float* __restrict__ p, int n) {
  int i = blockIdx.x * blockDim.x + threadIdx.x;
  if (i < n) p[i] = 0.0f;
}

// ---------------------------------------------------------------------------
// Kernel 3: edge conv (two-phase). One workgroup = 16 points of one batch.
// Weight matrix staged to LDS once via TDM; k-loop stages 16 bf16 edge-
// feature rows to LDS; WMMA bf16 GEMM; STATS pass accumulates per-channel
// sum/sumsq, apply pass does BN + LeakyReLU + running max over k.
// cat features live in bf16 (WMMA operand precision) to halve traffic.
// ---------------------------------------------------------------------------
template <int CIN, int CK, int COUT, int NWAVES, int OTPW, bool FIRST, bool STATS>
__global__ __launch_bounds__(NWAVES * 32)
void edgeconv_kernel(const float* __restrict__ xf,      // FIRST: (B,3,N) f32
                     const __bf16* __restrict__ xb,     // else: cat slice, bf16
                     int xstride,
                     const int* __restrict__ nbr,
                     const __bf16* __restrict__ Wb,
                     const float* __restrict__ coefA, const float* __restrict__ coefB,
                     float* __restrict__ gsum, float* __restrict__ gsq,
                     __bf16* __restrict__ out, int ostride, int N) {
  constexpr int NT = NWAVES * 32;
  constexpr int STRIDE = CK + 8;  // even elem stride -> 4B-aligned rows
  __shared__ alignas(16) __bf16 feat[16 * STRIDE];
  __shared__ alignas(16) __bf16 wlds[COUT * CK];
  __shared__ int sidx[16];
  __shared__ float ssum[COUT];
  __shared__ float ssq[COUT];

  const int tid = threadIdx.x;
  const int bpb = N >> 4;
  const int b = blockIdx.x / bpb;
  const int n0 = (blockIdx.x - b * bpb) << 4;
  const int lane = tid & 31, wave = tid >> 5;

  if (STATS) {
    for (int t = tid; t < COUT; t += NT) { ssum[t] = 0.0f; ssq[t] = 0.0f; }
  }
  // Stage W (COUT x CK bf16) into LDS with one TDM descriptor (wave 0 issues).
#if HAVE_TDM
  if (tid < 32) {
    tdm_load_2d(Wb, lds_offset_of(wlds), CK, COUT, CK, COUT,
                (unsigned long long)CK, /*2B*/1, 0, 0, 0);
    __builtin_amdgcn_s_wait_tensorcnt(0);
  }
#else
  for (int t = tid; t < COUT * CK; t += NT) wlds[t] = Wb[t];
#endif

  float rmax[OTPW][8];
  float lsum[OTPW], lsq[OTPW];
#pragma unroll
  for (int t = 0; t < OTPW; ++t) {
    lsum[t] = 0.0f; lsq[t] = 0.0f;
#pragma unroll
    for (int r = 0; r < 8; ++r) rmax[t][r] = -3.4e38f;
  }

  for (int k = 0; k < KNBR; ++k) {
    __syncthreads();  // publishes wlds/ssum on first pass; guards feat reuse
    if (tid < 16) sidx[tid] = nbr[((size_t)(b * N + n0 + tid)) * KNBR + k];
    __syncthreads();
    // Build 16 edge-feature rows: [x_j - x_n , x_n , 0-pad] in bf16.
    for (int t = tid; t < 16 * CK; t += NT) {
      int row = t / CK, c = t - row * CK;
      int n = n0 + row, j = sidx[row];
      float v = 0.0f;
      if (c < CIN) {
        float xjv = FIRST ? xf[((size_t)b * 3 + c) * N + j]
                          : (float)xb[((size_t)(b * N) + j) * xstride + c];
        float xnv = FIRST ? xf[((size_t)b * 3 + c) * N + n]
                          : (float)xb[((size_t)(b * N) + n) * xstride + c];
        v = xjv - xnv;
      } else if (c < 2 * CIN) {
        int c2 = c - CIN;
        v = FIRST ? xf[((size_t)b * 3 + c2) * N + n]
                  : (float)xb[((size_t)(b * N) + n) * xstride + c2];
      }
      feat[row * STRIDE + c] = (__bf16)v;
    }
    __syncthreads();

    floatx8 acc[OTPW];
#pragma unroll
    for (int t = 0; t < OTPW; ++t)
#pragma unroll
      for (int r = 0; r < 8; ++r) acc[t][r] = 0.0f;

    for (int kk = 0; kk < CK; kk += 32) {
      bf16x16 a = load_a_frag(feat, STRIDE, kk, lane);
#pragma unroll
      for (int t = 0; t < OTPW; ++t) {
        int o0 = (wave + t * NWAVES) << 4;
        bf16x16 bb = load_b_frag(wlds, CK, o0, kk, lane);
        acc[t] = __builtin_amdgcn_wmma_f32_16x16x32_bf16(
            false, a, false, bb, (short)0, acc[t], false, false);
      }
    }

    if (STATS) {
#pragma unroll
      for (int t = 0; t < OTPW; ++t)
#pragma unroll
        for (int r = 0; r < 8; ++r) {
          float v = acc[t][r];
          lsum[t] += v; lsq[t] += v * v;
        }
    } else {
#pragma unroll
      for (int t = 0; t < OTPW; ++t) {
        int ch = ((wave + t * NWAVES) << 4) + (lane & 15);
        float as = coefA[ch], bs = coefB[ch];
#pragma unroll
        for (int r = 0; r < 8; ++r) {
          float v = acc[t][r] * as + bs;
          v = (v >= 0.0f) ? v : NEG_SLOPE * v;
          rmax[t][r] = fmaxf(rmax[t][r], v);
        }
      }
    }
  }

  if (STATS) {
#pragma unroll
    for (int t = 0; t < OTPW; ++t) {
      int ch = ((wave + t * NWAVES) << 4) + (lane & 15);
      atomicAdd(&ssum[ch], lsum[t]);
      atomicAdd(&ssq[ch], lsq[t]);
    }
    __syncthreads();
    for (int t = tid; t < COUT; t += NT) {
      atomicAdd(&gsum[t], ssum[t]);
      atomicAdd(&gsq[t], ssq[t]);
    }
  } else {
    const int col = lane & 15, rofs = (lane >> 4) << 3;
#pragma unroll
    for (int t = 0; t < OTPW; ++t) {
      int o0 = (wave + t * NWAVES) << 4;
#pragma unroll
      for (int r = 0; r < 8; ++r) {
        int n = n0 + r + rofs;
        out[((size_t)(b * N) + n) * ostride + o0 + col] = (__bf16)rmax[t][r];
      }
    }
  }
}

// ---------------------------------------------------------------------------
// Kernel 4: conv5 (cat bf16 (512) -> 1024). Activation tile staged to LDS by
// one TDM descriptor (16 rows x 512 bf16, pad 4 DWORDs per 256 DWORDs ->
// LDS row stride 520). Apply pass reduces max/mean per tile in LDS, then one
// atomic per (b,channel) per workgroup.
// ---------------------------------------------------------------------------
template <bool STATS>
__global__ __launch_bounds__(256)
void conv5_kernel(const __bf16* __restrict__ catb, const __bf16* __restrict__ Wb,
                  const float* __restrict__ coefA, const float* __restrict__ coefB,
                  float* __restrict__ gsum, float* __restrict__ gsq,
                  unsigned* __restrict__ maxenc, float* __restrict__ sumbuf, int N) {
  constexpr int CK = 512, COUT = 1024, NWAVES = 8, OTPW = 8, NT = 256;
  constexpr int STRIDE = CK + 8;
  __shared__ alignas(16) __bf16 feat[16 * STRIDE];
  __shared__ float s0[COUT];  // STATS: sum   | apply: per-tile max (enc bits)
  __shared__ float s1[COUT];  // STATS: sumsq | apply: per-tile sum

  const int tid = threadIdx.x;
  const int bpb = N >> 4;
  const int b = blockIdx.x / bpb;
  const int n0 = (blockIdx.x - b * bpb) << 4;
  const int lane = tid & 31, wave = tid >> 5;

  for (int t = tid; t < COUT; t += NT) { s0[t] = 0.0f; s1[t] = 0.0f; }
#if HAVE_TDM
  if (tid < 32) {
    tdm_load_2d(catb + (size_t)(b * N + n0) * CK, lds_offset_of(feat),
                CK, 16, CK, 16, (unsigned long long)CK, /*2B*/1,
                /*pad_en*/1, /*every 256 DWORDs*/7, /*+4 DWORDs*/3);
    __builtin_amdgcn_s_wait_tensorcnt(0);
  }
#else
  for (int t = tid; t < 16 * CK; t += NT) {
    int row = t / CK, c = t - row * CK;
    feat[row * STRIDE + c] = catb[((size_t)(b * N + n0 + row)) * CK + c];
  }
#endif
  // Each wave prefetches the head of its weight panel (L2 temporal-hint path).
  __builtin_prefetch(Wb + (size_t)(wave << 4) * CK, 0, 1);
  __syncthreads();

  floatx8 acc[OTPW];
#pragma unroll
  for (int t = 0; t < OTPW; ++t)
#pragma unroll
    for (int r = 0; r < 8; ++r) acc[t][r] = 0.0f;

  for (int kk = 0; kk < CK; kk += 32) {
    bf16x16 a = load_a_frag(feat, STRIDE, kk, lane);
#pragma unroll
    for (int t = 0; t < OTPW; ++t) {
      int o0 = (wave + t * NWAVES) << 4;
      bf16x16 bb = load_b_frag(Wb, CK, o0, kk, lane);
      acc[t] = __builtin_amdgcn_wmma_f32_16x16x32_bf16(
          false, a, false, bb, (short)0, acc[t], false, false);
    }
  }

  if (STATS) {
#pragma unroll
    for (int t = 0; t < OTPW; ++t) {
      int ch = ((wave + t * NWAVES) << 4) + (lane & 15);
      float ls = 0.0f, lq = 0.0f;
#pragma unroll
      for (int r = 0; r < 8; ++r) { float v = acc[t][r]; ls += v; lq += v * v; }
      atomicAdd(&s0[ch], ls);
      atomicAdd(&s1[ch], lq);
    }
    __syncthreads();
    for (int t = tid; t < COUT; t += NT) {
      atomicAdd(&gsum[t], s0[t]);
      atomicAdd(&gsq[t], s1[t]);
    }
  } else {
#pragma unroll
    for (int t = 0; t < OTPW; ++t) {
      int ch = ((wave + t * NWAVES) << 4) + (lane & 15);
      float as = coefA[ch], bs = coefB[ch];
      float ls = 0.0f;
      unsigned lm = 0u;
#pragma unroll
      for (int r = 0; r < 8; ++r) {
        float v = acc[t][r] * as + bs;
        v = (v >= 0.0f) ? v : NEG_SLOPE * v;
        ls += v;
        unsigned e = fenc(v);
        lm = (e > lm) ? e : lm;
      }
      atomicMax(reinterpret_cast<unsigned*>(&s0[ch]), lm);
      atomicAdd(&s1[ch], ls);
    }
    __syncthreads();
    for (int t = tid; t < COUT; t += NT) {
      atomicMax(&maxenc[(size_t)b * COUT + t], __float_as_uint(s0[t]));
      atomicAdd(&sumbuf[(size_t)b * COUT + t], s1[t]);
    }
  }
}

// ---------------------------------------------------------------------------
// Small kernels: BN coefficient finalize, z assembly, FC, batch-BN.
// ---------------------------------------------------------------------------
__global__ void finalize_stats(const float* __restrict__ gsum, const float* __restrict__ gsq,
                               const float* __restrict__ gamma, const float* __restrict__ beta,
                               float* __restrict__ coefA, float* __restrict__ coefB,
                               float invcount, int C) {
  int c = blockIdx.x * blockDim.x + threadIdx.x;
  if (c >= C) return;
  float mean = gsum[c] * invcount;
  float var = gsq[c] * invcount - mean * mean;
  float rstd = rsqrtf(var + BN_EPS);
  float a = gamma[c] * rstd;
  coefA[c] = a;
  coefB[c] = beta[c] - mean * a;
}

__global__ void build_z(const unsigned* __restrict__ maxenc, const float* __restrict__ sumbuf,
                        float* __restrict__ z) {
  int i = blockIdx.x * blockDim.x + threadIdx.x;
  if (i >= BATCH * 2048) return;
  int b = i >> 11, c = i & 2047;
  z[i] = (c < 1024) ? fdec(maxenc[(size_t)b * 1024 + c])
                    : sumbuf[(size_t)b * 1024 + (c - 1024)] * (1.0f / NPTS);
}

__global__ void fc_kernel(const float* __restrict__ in, const float* __restrict__ W,
                          const float* __restrict__ bias, float* __restrict__ out,
                          int B, int I, int O) {
  int i = blockIdx.x * blockDim.x + threadIdx.x;
  if (i >= B * O) return;
  int b = i / O, o = i - b * O;
  const float* ir = in + (size_t)b * I;
  const float* wr = W + (size_t)o * I;
  float s = bias ? bias[o] : 0.0f;
#pragma unroll 8
  for (int c = 0; c < I; ++c) s += ir[c] * wr[c];
  out[i] = s;
}

__global__ void bnrow_kernel(const float* __restrict__ in, const float* __restrict__ g,
                             const float* __restrict__ bt, float* __restrict__ out, int O) {
  int c = blockIdx.x * blockDim.x + threadIdx.x;
  if (c >= O) return;
  float s = 0.0f, q = 0.0f;
  for (int b = 0; b < BATCH; ++b) { float v = in[(size_t)b * O + c]; s += v; q += v * v; }
  float mean = s * (1.0f / BATCH);
  float var = q * (1.0f / BATCH) - mean * mean;
  float rstd = rsqrtf(var + BN_EPS);
  float a = g[c] * rstd, bb = bt[c] - mean * a;
  for (int b = 0; b < BATCH; ++b) {
    float v = in[(size_t)b * O + c] * a + bb;
    out[(size_t)b * O + c] = (v >= 0.0f) ? v : NEG_SLOPE * v;
  }
}

// ---------------------------------------------------------------------------
// Host-side orchestration
// ---------------------------------------------------------------------------
extern "C" void kernel_launch(void* const* d_in, const int* in_sizes, int n_in,
                              void* d_out, int out_size, void* d_ws, size_t ws_size,
                              hipStream_t stream) {
  (void)in_sizes; (void)n_in; (void)out_size; (void)ws_size;
  const float* x    = (const float*)d_in[0];   // (8,3,2048)
  const float* geod = (const float*)d_in[1];   // (8,2048,2048)
  const float* W1 = (const float*)d_in[2];  const float* g1 = (const float*)d_in[3];  const float* bt1 = (const float*)d_in[4];
  const float* W2 = (const float*)d_in[5];  const float* g2 = (const float*)d_in[6];  const float* bt2 = (const float*)d_in[7];
  const float* W3 = (const float*)d_in[8];  const float* g3 = (const float*)d_in[9];  const float* bt3 = (const float*)d_in[10];
  const float* W4 = (const float*)d_in[11]; const float* g4 = (const float*)d_in[12]; const float* bt4 = (const float*)d_in[13];
  const float* W5 = (const float*)d_in[14]; const float* g5 = (const float*)d_in[15]; const float* bt5 = (const float*)d_in[16];
  const float* L1 = (const float*)d_in[17]; const float* g6 = (const float*)d_in[18]; const float* bt6 = (const float*)d_in[19];
  const float* L2 = (const float*)d_in[20]; const float* bl2 = (const float*)d_in[21];
  const float* g7 = (const float*)d_in[22]; const float* bt7 = (const float*)d_in[23];
  const float* L3 = (const float*)d_in[24]; const float* bl3 = (const float*)d_in[25];
  float* outp = (float*)d_out;

  // ---- carve workspace ----
  char* w = (char*)d_ws;
  size_t off = 0;
  auto alloc = [&](size_t bytes) -> void* {
    void* p = w + off;
    off = (off + bytes + 255) & ~(size_t)255;
    return p;
  };
  int*     idx    = (int*)alloc((size_t)BATCH * NPTS * KNBR * 4);
  __bf16*  cat    = (__bf16*)alloc((size_t)BATCH * NPTS * 512 * 2);
  __bf16*  w1b    = (__bf16*)alloc((size_t)64 * 32 * 2);
  __bf16*  w2b    = (__bf16*)alloc((size_t)64 * 128 * 2);
  __bf16*  w3b    = (__bf16*)alloc((size_t)128 * 128 * 2);
  __bf16*  w4b    = (__bf16*)alloc((size_t)256 * 256 * 2);
  __bf16*  w5b    = (__bf16*)alloc((size_t)1024 * 512 * 2);
  float*   stats  = (float*)alloc((size_t)4 * 1024 * 4); // gsum|gsq|coefA|coefB
  float* gsum = stats, *gsq = stats + 1024, *coefA = stats + 2048, *coefB = stats + 3072;
  float*   pool   = (float*)alloc((size_t)2 * BATCH * 1024 * 4); // maxenc|sumbuf
  unsigned* maxenc = (unsigned*)pool;
  float*    sumbuf = pool + BATCH * 1024;
  float*   z      = (float*)alloc((size_t)BATCH * 2048 * 4);
  float*   z1     = (float*)alloc((size_t)BATCH * 512 * 4);
  float*   z2     = (float*)alloc((size_t)BATCH * 256 * 4);

  const int N = NPTS;
  const int grid16 = BATCH * (N / 16);  // 1024

  // ---- 1. neighbor indices ----
  topk_kernel<<<BATCH * N, 256, 0, stream>>>(geod, idx);

  // ---- 2. weight conversion to padded bf16 ----
  convw_kernel<<<(64 * 32 + 255) / 256, 256, 0, stream>>>(W1, w1b, 64, 6, 32);
  convw_kernel<<<(64 * 128 + 255) / 256, 256, 0, stream>>>(W2, w2b, 64, 128, 128);
  convw_kernel<<<(128 * 128 + 255) / 256, 256, 0, stream>>>(W3, w3b, 128, 128, 128);
  convw_kernel<<<(256 * 256 + 255) / 256, 256, 0, stream>>>(W4, w4b, 256, 256, 256);
  convw_kernel<<<(1024 * 512 + 255) / 256, 256, 0, stream>>>(W5, w5b, 1024, 512, 512);

  const float invEC = 1.0f / ((float)BATCH * N * KNBR);
  const float invBN = 1.0f / ((float)BATCH * N);

  // ---- 3. edge conv 1: (B,3,N) -> cat[:, :, 0:64] ----
  zero_f32<<<8, 256, 0, stream>>>(gsum, 2048);
  edgeconv_kernel<3, 32, 64, 4, 1, true, true><<<grid16, 128, 0, stream>>>(
      x, nullptr, 0, idx, w1b, coefA, coefB, gsum, gsq, nullptr, 0, N);
  finalize_stats<<<1, 64, 0, stream>>>(gsum, gsq, g1, bt1, coefA, coefB, invEC, 64);
  edgeconv_kernel<3, 32, 64, 4, 1, true, false><<<grid16, 128, 0, stream>>>(
      x, nullptr, 0, idx, w1b, coefA, coefB, gsum, gsq, cat + 0, 512, N);

  // ---- 4. edge conv 2: cat[0:64] -> cat[64:128] ----
  zero_f32<<<8, 256, 0, stream>>>(gsum, 2048);
  edgeconv_kernel<64, 128, 64, 4, 1, false, true><<<grid16, 128, 0, stream>>>(
      nullptr, cat + 0, 512, idx, w2b, coefA, coefB, gsum, gsq, nullptr, 0, N);
  finalize_stats<<<1, 64, 0, stream>>>(gsum, gsq, g2, bt2, coefA, coefB, invEC, 64);
  edgeconv_kernel<64, 128, 64, 4, 1, false, false><<<grid16, 128, 0, stream>>>(
      nullptr, cat + 0, 512, idx, w2b, coefA, coefB, gsum, gsq, cat + 64, 512, N);

  // ---- 5. edge conv 3: cat[64:128] -> cat[128:256] ----
  zero_f32<<<8, 256, 0, stream>>>(gsum, 2048);
  edgeconv_kernel<64, 128, 128, 8, 1, false, true><<<grid16, 256, 0, stream>>>(
      nullptr, cat + 64, 512, idx, w3b, coefA, coefB, gsum, gsq, nullptr, 0, N);
  finalize_stats<<<1, 128, 0, stream>>>(gsum, gsq, g3, bt3, coefA, coefB, invEC, 128);
  edgeconv_kernel<64, 128, 128, 8, 1, false, false><<<grid16, 256, 0, stream>>>(
      nullptr, cat + 64, 512, idx, w3b, coefA, coefB, gsum, gsq, cat + 128, 512, N);

  // ---- 6. edge conv 4: cat[128:256] -> cat[256:512] ----
  zero_f32<<<8, 256, 0, stream>>>(gsum, 2048);
  edgeconv_kernel<128, 256, 256, 8, 2, false, true><<<grid16, 256, 0, stream>>>(
      nullptr, cat + 128, 512, idx, w4b, coefA, coefB, gsum, gsq, nullptr, 0, N);
  finalize_stats<<<1, 256, 0, stream>>>(gsum, gsq, g4, bt4, coefA, coefB, invEC, 256);
  edgeconv_kernel<128, 256, 256, 8, 2, false, false><<<grid16, 256, 0, stream>>>(
      nullptr, cat + 128, 512, idx, w4b, coefA, coefB, gsum, gsq, cat + 256, 512, N);

  // ---- 7. conv5: cat(512) -> 1024, BN, lrelu, max+mean over N ----
  zero_f32<<<8, 256, 0, stream>>>(gsum, 2048);
  conv5_kernel<true><<<grid16, 256, 0, stream>>>(
      cat, w5b, coefA, coefB, gsum, gsq, maxenc, sumbuf, N);
  finalize_stats<<<4, 256, 0, stream>>>(gsum, gsq, g5, bt5, coefA, coefB, invBN, 1024);
  zero_f32<<<(2 * BATCH * 1024 + 255) / 256, 256, 0, stream>>>(pool, 2 * BATCH * 1024);
  conv5_kernel<false><<<grid16, 256, 0, stream>>>(
      cat, w5b, coefA, coefB, gsum, gsq, maxenc, sumbuf, N);

  // ---- 8. z = [max, mean] -> FC stack ----
  build_z<<<(BATCH * 2048 + 255) / 256, 256, 0, stream>>>(maxenc, sumbuf, z);

  fc_kernel<<<(BATCH * 512 + 255) / 256, 256, 0, stream>>>(z, L1, nullptr, z1, BATCH, 2048, 512);
  bnrow_kernel<<<2, 256, 0, stream>>>(z1, g6, bt6, z1, 512);

  fc_kernel<<<(BATCH * 256 + 255) / 256, 256, 0, stream>>>(z1, L2, bl2, z2, BATCH, 512, 256);
  bnrow_kernel<<<1, 256, 0, stream>>>(z2, g7, bt7, z2, 256);

  fc_kernel<<<2, 256, 0, stream>>>(z2, L3, bl3, outp, BATCH, 256, 40);
}
